// Net_10256381903067
// MI455X (gfx1250) — matrix operand order repaired and checked by reading, not compiled, not run on hardware
//
#include <hip/hip_runtime.h>
#include <math.h>

typedef __attribute__((ext_vector_type(16))) _Float16 v16h;
typedef __attribute__((ext_vector_type(8)))  _Float16 v8h;
typedef __attribute__((ext_vector_type(8)))  float    v8f;

#define DIM 32

__device__ __forceinline__ float sigf(float x) { return 1.0f / (1.0f + expf(-x)); }

// order-preserving float <-> uint encoding for atomicMax
__device__ __forceinline__ unsigned encf(float f) {
    unsigned u = __float_as_uint(f);
    return (u & 0x80000000u) ? ~u : (u | 0x80000000u);
}
__device__ __forceinline__ float decf(unsigned u) {
    return (u & 0x80000000u) ? __uint_as_float(u & 0x7FFFFFFFu) : __uint_as_float(~u);
}

// ---------------- degree: deg[d] += 1 over edges ----------------
__global__ void k_deg(const int* __restrict__ ei, float* __restrict__ deg, int E) {
    int e = blockIdx.x * blockDim.x + threadIdx.x;
    if (e < E) atomicAdd(&deg[ei[E + e]], 1.0f);
}

// ---------------- out = relu(x @ lin0_w + lin0_b) ----------------
__global__ void k_lin0(const float* __restrict__ x, const float* __restrict__ w,
                       const float* __restrict__ b, float* __restrict__ out, int N) {
    int t = blockIdx.x * blockDim.x + threadIdx.x;
    if (t >= N * DIM) return;
    int n = t >> 5, j = t & 31;
    out[t] = fmaxf(x[n] * w[j] + b[j], 0.0f);
}

// ---------------- W2 (128,1024) f32 -> W2^T (1024,128) f16 ----------------
__global__ void k_w2t(const float* __restrict__ w2, _Float16* __restrict__ w2t) {
    int t = blockIdx.x * blockDim.x + threadIdx.x;
    if (t >= 128 * 1024) return;
    int k = t >> 10, n = t & 1023;
    w2t[n * 128 + k] = (_Float16)w2[t];
}

// ---------------- h2 = relu(edge_attr @ W1 + b1), f16 (Epad x 128) ----------------
__global__ void k_h2(const float* __restrict__ ea, const float* __restrict__ w1,
                     const float* __restrict__ b1, _Float16* __restrict__ h2,
                     int E, int Epad) {
    int g = blockIdx.x * blockDim.x + threadIdx.x;
    int e = g >> 5, lane = g & 31;
    if (e >= Epad) return;
    float a0 = 0.f, a1 = 0.f;
    if (e < E) { a0 = ea[2 * e]; a1 = ea[2 * e + 1]; }
    for (int q = 0; q < 4; ++q) {
        int k = lane + 32 * q;
        float h = a0 * w1[k] + a1 * w1[128 + k] + b1[k];
        h2[(size_t)e * 128 + k] = (_Float16)fmaxf(h, 0.0f);
    }
}

// ---------------- WMMA GEMM: ew = h2(Ex128) @ W2(128x1024) + b2, f16 out ----------------
// one wave per 16x16 output tile; K = 128 = 4 x (16x16x32 f16 WMMA)
__global__ void k_gemm_ew(const _Float16* __restrict__ h2, const _Float16* __restrict__ w2t,
                          const float* __restrict__ b2, _Float16* __restrict__ ew) {
    int lane  = threadIdx.x & 31;
    int wv    = threadIdx.x >> 5;          // 0..3
    int etile = blockIdx.x;
    int ntile = blockIdx.y * 4 + wv;       // 0..63
    int row = lane & 15, hi = lane >> 4;

    // A fragment (16x32 f16): lane<16 -> row, K 0-7 & 16-23 ; lane>=16 -> row, K 8-15 & 24-31
    const _Float16* arow = h2 + (size_t)(etile * 16 + row) * 128 + hi * 8;
    // B fragment (32x16 f16): lane<16 -> col=lane, K 0-7 & 16-23 ; lane>=16 -> col=lane-16, K 8-15 & 24-31
    const _Float16* brow = w2t + (size_t)(ntile * 16 + row) * 128 + hi * 8;

    v8f c = {};
    for (int kt = 0; kt < 4; ++kt) {
        v8h alo = *(const v8h*)(arow + kt * 32);
        v8h ahi = *(const v8h*)(arow + kt * 32 + 16);
        v8h bl  = *(const v8h*)(brow + kt * 32);
        v8h bh  = *(const v8h*)(brow + kt * 32 + 16);
        v16h a, b;
        for (int j = 0; j < 8; ++j) {
            a[j] = alo[j]; a[8 + j] = ahi[j];
            b[j] = bl[j];  b[8 + j] = bh[j];
        }
        c = __builtin_amdgcn_wmma_f32_16x16x32_f16(false, a, false, b, (short)0, c, false, false);
    }
    // C/D layout: lane<16 -> N=lane, M=vgpr r ; lane>=16 -> N=lane-16, M=8+r
    int col = ntile * 16 + row;
    float bias = b2[col];
    size_t base = (size_t)(etile * 16 + hi * 8) * 1024 + col;
    for (int r = 0; r < 8; ++r)
        ew[base + (size_t)r * 1024] = (_Float16)(c[r] + bias);
}

// ---------------- msg[e] = out[src] @ ew[e]; scatter-add into agg[dst] ----------------
__global__ void k_msg(const float* __restrict__ out, const _Float16* __restrict__ ew,
                      const int* __restrict__ ei, float* __restrict__ agg, int E) {
    int g = blockIdx.x * blockDim.x + threadIdx.x;
    int e = g >> 5, j = g & 31;
    if (e >= E) return;
    int s = ei[e], d = ei[E + e];
    float xs = out[(size_t)s * DIM + j];
    const _Float16* W = ew + (size_t)e * 1024;
    float acc = 0.0f;
    for (int i = 0; i < DIM; ++i) {
        float xi = __shfl(xs, i, 32);
        acc += xi * (float)W[i * 32 + j];
    }
    atomicAdd(&agg[(size_t)d * DIM + j], acc);
}

// ---------------- node update: NNConv root+bias+relu, then one GRU step ----------------
__global__ void k_node(float* __restrict__ out, const float* __restrict__ agg,
                       const float* __restrict__ deg, const float* __restrict__ rootw,
                       const float* __restrict__ cbias, const float* __restrict__ wih,
                       const float* __restrict__ whh, const float* __restrict__ bih,
                       const float* __restrict__ bhh, int N) {
    int g = blockIdx.x * blockDim.x + threadIdx.x;
    int n = g >> 5, j = g & 31;
    if (n >= N) return;
    float h = out[(size_t)n * DIM + j];
    float r = 0.0f;
    for (int i = 0; i < DIM; ++i) r += __shfl(h, i, 32) * rootw[i * DIM + j];
    float m = agg[(size_t)n * DIM + j] / fmaxf(deg[n], 1.0f) + r + cbias[j];
    m = fmaxf(m, 0.0f);
    float gi0 = bih[j], gi1 = bih[j + 32], gi2 = bih[j + 64];
    float gh0 = bhh[j], gh1 = bhh[j + 32], gh2 = bhh[j + 64];
    for (int i = 0; i < DIM; ++i) {
        float mi = __shfl(m, i, 32), hi = __shfl(h, i, 32);
        gi0 += mi * wih[i * 96 + j];
        gi1 += mi * wih[i * 96 + j + 32];
        gi2 += mi * wih[i * 96 + j + 64];
        gh0 += hi * whh[i * 96 + j];
        gh1 += hi * whh[i * 96 + j + 32];
        gh2 += hi * whh[i * 96 + j + 64];
    }
    float rg = sigf(gi0 + gh0), zg = sigf(gi1 + gh1);
    float ng = tanhf(gi2 + rg * gh2);
    out[(size_t)n * DIM + j] = (1.0f - zg) * ng + zg * h;
}

// ---------------- Set2Set LSTM step (one wave per graph) ----------------
__global__ void k_lstm(const float* __restrict__ qs, float* __restrict__ hh,
                       float* __restrict__ cc, float* __restrict__ qv,
                       const float* __restrict__ wih, const float* __restrict__ whh,
                       const float* __restrict__ bih, const float* __restrict__ bhh,
                       unsigned* __restrict__ emax, float* __restrict__ asum, int B) {
    int g = blockIdx.x * blockDim.x + threadIdx.x;
    int b = g >> 5, j = g & 31;
    if (b >= B) return;
    float g0 = bih[j] + bhh[j];
    float g1 = bih[j + 32] + bhh[j + 32];
    float g2 = bih[j + 64] + bhh[j + 64];
    float g3 = bih[j + 96] + bhh[j + 96];
    for (int k = 0; k < 64; ++k) {
        float q = qs[b * 64 + k];
        g0 += q * wih[k * 128 + j];
        g1 += q * wih[k * 128 + j + 32];
        g2 += q * wih[k * 128 + j + 64];
        g3 += q * wih[k * 128 + j + 96];
    }
    for (int k = 0; k < 32; ++k) {
        float hv = hh[b * 32 + k];
        g0 += hv * whh[k * 128 + j];
        g1 += hv * whh[k * 128 + j + 32];
        g2 += hv * whh[k * 128 + j + 64];
        g3 += hv * whh[k * 128 + j + 96];
    }
    float cn = sigf(g1) * cc[b * 32 + j] + sigf(g0) * tanhf(g2);  // f*c + i*g
    float hn = sigf(g3) * tanhf(cn);                              // o*tanh(c)
    cc[b * 32 + j] = cn;
    hh[b * 32 + j] = hn;
    qv[b * 32 + j] = hn;
    if (j == 0) { emax[b] = 0u; asum[b] = 0.0f; }  // 0 < encf(any finite float)
}

// ---------------- attention logits e[n] = <out[n], q[batch[n]]>, seg-max ----------------
__global__ void k_att_e(const float* __restrict__ out, const float* __restrict__ qv,
                        const int* __restrict__ batch, float* __restrict__ ebuf,
                        unsigned* __restrict__ emax, int N) {
    int g = blockIdx.x * blockDim.x + threadIdx.x;
    int n = g >> 5, j = g & 31;
    if (n >= N) return;
    int b = batch[n];
    float p = out[(size_t)n * DIM + j] * qv[b * DIM + j];
    for (int off = 16; off; off >>= 1) p += __shfl_xor(p, off, 32);
    if (j == 0) {
        ebuf[n] = p;
        atomicMax(&emax[b], encf(p));
    }
}

// ---------------- a[n] = exp(e - max); seg-sum ----------------
__global__ void k_att_a(const float* __restrict__ ebuf, const unsigned* __restrict__ emax,
                        const int* __restrict__ batch, float* __restrict__ abuf,
                        float* __restrict__ asum, int N) {
    int n = blockIdx.x * blockDim.x + threadIdx.x;
    if (n >= N) return;
    int b = batch[n];
    float a = expf(ebuf[n] - decf(emax[b]));
    abuf[n] = a;
    atomicAdd(&asum[b], a);
}

// ---------------- rvec[b] += (a/asum) * out[n] ----------------
__global__ void k_att_r(const float* __restrict__ out, const float* __restrict__ abuf,
                        const float* __restrict__ asum, const int* __restrict__ batch,
                        float* __restrict__ rvec, int N) {
    int g = blockIdx.x * blockDim.x + threadIdx.x;
    int n = g >> 5, j = g & 31;
    if (n >= N) return;
    int b = batch[n];
    float w = abuf[n] / asum[b];
    atomicAdd(&rvec[b * DIM + j], w * out[(size_t)n * DIM + j]);
}

// ---------------- q_star = concat(q, rvec) ----------------
__global__ void k_qstar(const float* __restrict__ qv, const float* __restrict__ rvec,
                        float* __restrict__ qs, int B) {
    int t = blockIdx.x * blockDim.x + threadIdx.x;
    if (t >= B * 64) return;
    int b = t >> 6, j = t & 63;
    qs[t] = (j < 32) ? qv[b * 32 + j] : rvec[b * 32 + (j - 32)];
}

// ---------------- head: y = (relu(qs@lin1+b1))@lin2 + b2 ----------------
__global__ void k_final(const float* __restrict__ qs, const float* __restrict__ l1w,
                        const float* __restrict__ l1b, const float* __restrict__ l2w,
                        const float* __restrict__ l2b, float* __restrict__ y, int B) {
    int g = blockIdx.x * blockDim.x + threadIdx.x;
    int b = g >> 5, j = g & 31;
    if (b >= B) return;
    float t = l1b[j];
    for (int k = 0; k < 64; ++k) t += qs[b * 64 + k] * l1w[k * 32 + j];
    t = fmaxf(t, 0.0f);
    float p = t * l2w[j];
    for (int off = 16; off; off >>= 1) p += __shfl_xor(p, off, 32);
    if (j == 0) y[b] = p + l2b[0];
}

extern "C" void kernel_launch(void* const* d_in, const int* in_sizes, int n_in,
                              void* d_out, int out_size, void* d_ws, size_t ws_size,
                              hipStream_t stream) {
    const float* x       = (const float*)d_in[0];
    const float* eattr   = (const float*)d_in[1];
    const float* lin0_w  = (const float*)d_in[2];
    const float* lin0_b  = (const float*)d_in[3];
    const float* nn_w1   = (const float*)d_in[4];
    const float* nn_b1   = (const float*)d_in[5];
    const float* nn_w2   = (const float*)d_in[6];
    const float* nn_b2   = (const float*)d_in[7];
    const float* rootw   = (const float*)d_in[8];
    const float* cbias   = (const float*)d_in[9];
    const float* gwih    = (const float*)d_in[10];
    const float* gwhh    = (const float*)d_in[11];
    const float* gbih    = (const float*)d_in[12];
    const float* gbhh    = (const float*)d_in[13];
    const float* lwih    = (const float*)d_in[14];
    const float* lwhh    = (const float*)d_in[15];
    const float* lbih    = (const float*)d_in[16];
    const float* lbhh    = (const float*)d_in[17];
    const float* l1w     = (const float*)d_in[18];
    const float* l1b     = (const float*)d_in[19];
    const float* l2w     = (const float*)d_in[20];
    const float* l2b     = (const float*)d_in[21];
    const int*   eidx    = (const int*)d_in[22];
    const int*   batch   = (const int*)d_in[23];
    float* y = (float*)d_out;

    const int N = in_sizes[0];        // 10000
    const int E = in_sizes[1] / 2;    // 160000
    const int B = out_size;           // 64
    const int Epad = (E + 15) & ~15;

    // workspace layout (256B aligned blocks)
    char* ws = (char*)d_ws;
    size_t off = 0;
    auto alloc = [&](size_t bytes) { size_t o = off; off = (off + bytes + 255) & ~(size_t)255; return o; };
    _Float16* ew   = (_Float16*)(ws + alloc((size_t)Epad * 1024 * 2));
    _Float16* h2   = (_Float16*)(ws + alloc((size_t)Epad * 128 * 2));
    _Float16* w2t  = (_Float16*)(ws + alloc((size_t)1024 * 128 * 2));
    float*    deg  = (float*)(ws + alloc((size_t)N * 4));
    float*    out  = (float*)(ws + alloc((size_t)N * DIM * 4));
    float*    agg  = (float*)(ws + alloc((size_t)N * DIM * 4));
    float*    qs   = (float*)(ws + alloc((size_t)B * 64 * 4));
    float*    hh   = (float*)(ws + alloc((size_t)B * DIM * 4));
    float*    cc   = (float*)(ws + alloc((size_t)B * DIM * 4));
    float*    qv   = (float*)(ws + alloc((size_t)B * DIM * 4));
    float*    rvec = (float*)(ws + alloc((size_t)B * DIM * 4));
    float*    ebuf = (float*)(ws + alloc((size_t)N * 4));
    float*    abuf = (float*)(ws + alloc((size_t)N * 4));
    unsigned* emax = (unsigned*)(ws + alloc((size_t)B * 4));
    float*    asum = (float*)(ws + alloc((size_t)B * 4));
    (void)ws_size; (void)n_in;

    const int T = 256;
    auto cdiv = [](int a, int b) { return (a + b - 1) / b; };

    // ---- precompute: deg, lin0, W2^T(f16), h2(f16), ew = h2@W2 + b2 (WMMA) ----
    hipMemsetAsync(deg, 0, (size_t)N * 4, stream);
    k_deg<<<cdiv(E, T), T, 0, stream>>>(eidx, deg, E);
    k_lin0<<<cdiv(N * DIM, T), T, 0, stream>>>(x, lin0_w, lin0_b, out, N);
    k_w2t<<<cdiv(128 * 1024, T), T, 0, stream>>>(nn_w2, w2t);
    k_h2<<<cdiv(Epad * 32, T), T, 0, stream>>>(eattr, nn_w1, nn_b1, h2, E, Epad);
    k_gemm_ew<<<dim3(Epad / 16, 16), 128, 0, stream>>>(h2, w2t, nn_b2, ew);

    // ---- 3x (NNConv message passing + GRU) ----
    for (int it = 0; it < 3; ++it) {
        hipMemsetAsync(agg, 0, (size_t)N * DIM * 4, stream);
        k_msg<<<cdiv(E * 32, T), T, 0, stream>>>(out, ew, eidx, agg, E);
        k_node<<<cdiv(N * 32, T), T, 0, stream>>>(out, agg, deg, rootw, cbias,
                                                  gwih, gwhh, gbih, gbhh, N);
    }

    // ---- Set2Set (3 steps) ----
    hipMemsetAsync(qs, 0, (size_t)B * 64 * 4, stream);
    hipMemsetAsync(hh, 0, (size_t)B * DIM * 4, stream);
    hipMemsetAsync(cc, 0, (size_t)B * DIM * 4, stream);
    for (int s = 0; s < 3; ++s) {
        hipMemsetAsync(rvec, 0, (size_t)B * DIM * 4, stream);
        k_lstm<<<cdiv(B * 32, T), T, 0, stream>>>(qs, hh, cc, qv, lwih, lwhh,
                                                  lbih, lbhh, emax, asum, B);
        k_att_e<<<cdiv(N * 32, T), T, 0, stream>>>(out, qv, batch, ebuf, emax, N);
        k_att_a<<<cdiv(N, T), T, 0, stream>>>(ebuf, emax, batch, abuf, asum, N);
        k_att_r<<<cdiv(N * 32, T), T, 0, stream>>>(out, abuf, asum, batch, rvec, N);
        k_qstar<<<cdiv(B * 64, T), T, 0, stream>>>(qv, rvec, qs, B);
    }

    // ---- head ----
    k_final<<<cdiv(B * 32, T), T, 0, stream>>>(qs, l1w, l1b, l2w, l2b, y, B);
}
